// TA_32040456028260
// MI455X (gfx1250) — compile-verified, matrix operand order checked by reading, not run on hardware
//
#include <hip/hip_runtime.h>
#include <math.h>

#define T_DIM  128
#define BATCH  512
#define D_DATA 512
#define H_HID  512
#define G4H    (4 * H_HID)
#define EPS_CS 1e-8f
#define NB_RED 64   // reduction blocks per timestep in attention stats

typedef _Float16 f16;
typedef __attribute__((ext_vector_type(8)))  f16   v8h;
typedef __attribute__((ext_vector_type(16))) f16   v16h;
typedef __attribute__((ext_vector_type(8)))  float v8f;

__device__ __forceinline__ float sigmoidf(float x) {
    return 1.0f / (1.0f + __expf(-x));
}

// Load one 16x32 f16 WMMA fragment from a row-major [rows][ld] f16 array.
// ISA 7.12.2 16-bit A layout: lanes 0-15 hold row M=lane, K = {0..7, 16..23};
// lanes 16-31 hold row M=lane-16, K = {8..15, 24..31}.  B uses the mirrored
// layout with lanes = N columns, which for gates = A * W^T means each lane
// reads row (n) of row-major W[n][k] with the same K packing.
__device__ __forceinline__ v16h load_frag_f16(const f16* __restrict__ base,
                                              int row0, int ld, int k0, int lane) {
    const int half = lane >> 4;           // 0: K {0..7,16..23}, 1: K {8..15,24..31}
    const int r    = row0 + (lane & 15);
    const f16* p   = base + (size_t)r * (size_t)ld + k0 + half * 8;
    v8h lo = *reinterpret_cast<const v8h*>(p);        // K +0..7   (or +8..15)
    v8h hi = *reinterpret_cast<const v8h*>(p + 16);   // K +16..23 (or +24..31)
    return __builtin_shufflevector(lo, hi, 0, 1, 2, 3, 4, 5, 6, 7,
                                           8, 9, 10, 11, 12, 13, 14, 15);
}

// ---------------------------------------------------------------------------
// fp32 -> f16 bulk convert (weights and X)
// ---------------------------------------------------------------------------
__global__ void cvt_f16_kernel(const float* __restrict__ src,
                               f16* __restrict__ dst, int n) {
    int i = blockIdx.x * blockDim.x + threadIdx.x;
    if (i < n) dst[i] = (f16)src[i];
}

// ---------------------------------------------------------------------------
// Init: combined bias, zero c / h buffers / hidden_tensor[0]
// ---------------------------------------------------------------------------
__global__ void init_kernel(const float* __restrict__ b_ih,
                            const float* __restrict__ b_hh,
                            float* __restrict__ bias,
                            float* __restrict__ c,
                            f16* __restrict__ hA,
                            f16* __restrict__ hB,
                            float* __restrict__ hidden0) {
    int i = blockIdx.x * blockDim.x + threadIdx.x;
    if (i < G4H) bias[i] = b_ih[i] + b_hh[i];
    if (i < BATCH * H_HID) {
        c[i]       = 0.0f;
        hA[i]      = (f16)0.0f;
        hB[i]      = (f16)0.0f;
        hidden0[i] = 0.0f;
    }
}

// ---------------------------------------------------------------------------
// One LSTM step, fully fused:
//   gates = x_{t-1} @ W_ih^T + h_{t-1} @ W_hh^T + bias ; c,h update ; store h_t
// Block = 128 threads (4 waves). Wave w computes gate w's 32x32 tile with
// 2x2 v_wmma_f32_16x16x32_f16 accumulators over K = D + H.
// Grid = (BATCH/32, H/32) = (16,16).
// ---------------------------------------------------------------------------
__global__ __launch_bounds__(128)
void lstm_step_kernel(const f16*   __restrict__ Xh,     // [B][T][D] f16
                      const f16*   __restrict__ Wih,    // [4H][D]  f16
                      const f16*   __restrict__ Whh,    // [4H][H]  f16
                      const float* __restrict__ bias,   // [4H]
                      const f16*   __restrict__ h_in,   // [B][H] f16 (h_{t-1})
                      f16*         __restrict__ h_out,  // [B][H] f16 (h_t)
                      float*       __restrict__ c,      // [B][H] f32 (in/out)
                      float*       __restrict__ hidden, // [T][B][H] f32
                      int t) {
    const int tid  = threadIdx.x;
    const int wave = tid >> 5;            // 0..3 == gate i,f,g,o
    const int lane = tid & 31;
    const int b0   = blockIdx.x * 32;     // batch-row tile
    const int n0   = blockIdx.y * 32;     // hidden-col tile

    v8f acc00 = {}, acc01 = {}, acc10 = {}, acc11 = {};

    const int gbase = wave * H_HID + n0;  // this wave's gate rows in W

    // ---- phase 1: x_{t-1} @ W_ih^T  (K = D_DATA) ----
    {
        const f16* xbase = Xh + (size_t)(t - 1) * D_DATA;   // row b -> + b*T*D
        #pragma unroll 4
        for (int k0 = 0; k0 < D_DATA; k0 += 32) {
            v16h a0 = load_frag_f16(xbase, b0,      T_DIM * D_DATA, k0, lane);
            v16h a1 = load_frag_f16(xbase, b0 + 16, T_DIM * D_DATA, k0, lane);
            v16h w0 = load_frag_f16(Wih, gbase,      D_DATA, k0, lane);
            v16h w1 = load_frag_f16(Wih, gbase + 16, D_DATA, k0, lane);
            acc00 = __builtin_amdgcn_wmma_f32_16x16x32_f16(false, a0, false, w0, (short)0, acc00, false, false);
            acc01 = __builtin_amdgcn_wmma_f32_16x16x32_f16(false, a0, false, w1, (short)0, acc01, false, false);
            acc10 = __builtin_amdgcn_wmma_f32_16x16x32_f16(false, a1, false, w0, (short)0, acc10, false, false);
            acc11 = __builtin_amdgcn_wmma_f32_16x16x32_f16(false, a1, false, w1, (short)0, acc11, false, false);
        }
    }

    // ---- phase 2: h_{t-1} @ W_hh^T  (K = H_HID) ----
    {
        #pragma unroll 4
        for (int k0 = 0; k0 < H_HID; k0 += 32) {
            v16h a0 = load_frag_f16(h_in, b0,      H_HID, k0, lane);
            v16h a1 = load_frag_f16(h_in, b0 + 16, H_HID, k0, lane);
            v16h w0 = load_frag_f16(Whh, gbase,      H_HID, k0, lane);
            v16h w1 = load_frag_f16(Whh, gbase + 16, H_HID, k0, lane);
            acc00 = __builtin_amdgcn_wmma_f32_16x16x32_f16(false, a0, false, w0, (short)0, acc00, false, false);
            acc01 = __builtin_amdgcn_wmma_f32_16x16x32_f16(false, a0, false, w1, (short)0, acc01, false, false);
            acc10 = __builtin_amdgcn_wmma_f32_16x16x32_f16(false, a1, false, w0, (short)0, acc10, false, false);
            acc11 = __builtin_amdgcn_wmma_f32_16x16x32_f16(false, a1, false, w1, (short)0, acc11, false, false);
        }
    }

    // ---- exchange gate tiles through LDS ----
    __shared__ float lds_g[4][32][32];
    {
        // C/D layout: acc[j] @ lane L -> row = j + 8*(L>=16), col = L&15
        const int half = lane >> 4;
        const int col  = lane & 15;
        #pragma unroll
        for (int j = 0; j < 8; ++j) {
            lds_g[wave][ 0 + half * 8 + j][ 0 + col] = acc00[j];
            lds_g[wave][ 0 + half * 8 + j][16 + col] = acc01[j];
            lds_g[wave][16 + half * 8 + j][ 0 + col] = acc10[j];
            lds_g[wave][16 + half * 8 + j][16 + col] = acc11[j];
        }
    }
    __syncthreads();

    // ---- elementwise LSTM update: 1024 elements / 128 threads = 8 each ----
    for (int e = tid; e < 32 * 32; e += 128) {
        const int m = e >> 5;          // row in tile (batch)
        const int n = e & 31;          // col in tile (hidden unit)
        const int b = b0 + m;
        const int hc = n0 + n;
        const float ig = sigmoidf(lds_g[0][m][n] + bias[0 * H_HID + hc]);
        const float fg = sigmoidf(lds_g[1][m][n] + bias[1 * H_HID + hc]);
        const float gg = tanhf   (lds_g[2][m][n] + bias[2 * H_HID + hc]);
        const float og = sigmoidf(lds_g[3][m][n] + bias[3 * H_HID + hc]);
        const size_t ci = (size_t)b * H_HID + hc;
        const float cn = fg * c[ci] + ig * gg;
        c[ci] = cn;
        const float hn = og * tanhf(cn);
        h_out[ci] = (f16)hn;
        hidden[(size_t)t * BATCH * H_HID + ci] = hn;
    }
}

// ---------------------------------------------------------------------------
// Attention stats: per (t, chunk) block partial dot(h_t, h_final) and |h_t|^2
// grid = (NB_RED, T_DIM), block = 256. Deterministic two-pass reduction.
// ---------------------------------------------------------------------------
__global__ __launch_bounds__(256)
void attn_partial_kernel(const float* __restrict__ hidden,
                         float* __restrict__ partials) {
    const int t   = blockIdx.y;   // 0..127 (127 itself gives |h_fin|^2 via dot)
    const int blk = blockIdx.x;   // 0..NB_RED-1
    const int CH  = (BATCH * H_HID) / NB_RED;   // 4096
    const float* ht = hidden + (size_t)t * BATCH * H_HID + (size_t)blk * CH;
    const float* hf = hidden + (size_t)(T_DIM - 1) * BATCH * H_HID + (size_t)blk * CH;
    float dot = 0.0f, sq = 0.0f;
    for (int i = threadIdx.x; i < CH; i += 256) {
        const float a = ht[i];
        dot += a * hf[i];
        sq  += a * a;
    }
    __shared__ float sd[256], ss[256];
    sd[threadIdx.x] = dot;
    ss[threadIdx.x] = sq;
    __syncthreads();
    for (int s = 128; s > 0; s >>= 1) {
        if (threadIdx.x < s) {
            sd[threadIdx.x] += sd[threadIdx.x + s];
            ss[threadIdx.x] += ss[threadIdx.x + s];
        }
        __syncthreads();
    }
    if (threadIdx.x == 0) {
        partials[(t * NB_RED + blk) * 2 + 0] = sd[0];
        partials[(t * NB_RED + blk) * 2 + 1] = ss[0];
    }
}

__global__ void attn_reduce_kernel(const float* __restrict__ partials,
                                   float* __restrict__ dots,
                                   float* __restrict__ sqs) {
    const int t = threadIdx.x;   // 1 block, 128 threads
    float d = 0.0f, s = 0.0f;
    for (int b = 0; b < NB_RED; ++b) {
        d += partials[(t * NB_RED + b) * 2 + 0];
        s += partials[(t * NB_RED + b) * 2 + 1];
    }
    dots[t] = d;
    sqs[t]  = s;
}

__global__ void attn_softmax_kernel(const float* __restrict__ dots,
                                    const float* __restrict__ sqs,
                                    float* __restrict__ alpha) {
    __shared__ float sm[128];
    const int t = threadIdx.x;   // 1 block, 128 threads
    const float nfin = fmaxf(sqrtf(sqs[T_DIM - 1]), EPS_CS);
    float s = -INFINITY;
    if (t < T_DIM - 1) {
        const float nt = fmaxf(sqrtf(sqs[t]), EPS_CS);
        s = dots[t] / (nt * nfin);
    }
    sm[t] = s;
    __syncthreads();
    for (int o = 64; o > 0; o >>= 1) {
        if (t < o) sm[t] = fmaxf(sm[t], sm[t + o]);
        __syncthreads();
    }
    const float mx = sm[0];
    __syncthreads();
    const float e = (t < T_DIM - 1) ? __expf(s - mx) : 0.0f;
    sm[t] = e;
    __syncthreads();
    for (int o = 64; o > 0; o >>= 1) {
        if (t < o) sm[t] += sm[t + o];
        __syncthreads();
    }
    if (t < T_DIM - 1) alpha[t] = e / sm[0];
}

__global__ __launch_bounds__(256)
void weighted_sum_kernel(const float* __restrict__ hidden,
                         const float* __restrict__ alpha,
                         float* __restrict__ out) {
    __shared__ float sa[128];
    if (threadIdx.x < T_DIM - 1) sa[threadIdx.x] = alpha[threadIdx.x];
    __syncthreads();
    const int i = blockIdx.x * 256 + threadIdx.x;   // over B*H
    float s = 0.0f;
    for (int t = 0; t < T_DIM - 1; ++t)
        s += sa[t] * hidden[(size_t)t * BATCH * H_HID + i];
    out[i] = s;
}

// ---------------------------------------------------------------------------
extern "C" void kernel_launch(void* const* d_in, const int* in_sizes, int n_in,
                              void* d_out, int out_size, void* d_ws, size_t ws_size,
                              hipStream_t stream) {
    const float* X    = (const float*)d_in[0];   // [B,T,D]
    const float* W_ih = (const float*)d_in[1];   // [4H,D]
    const float* W_hh = (const float*)d_in[2];   // [4H,H]
    const float* b_ih = (const float*)d_in[3];   // [4H]
    const float* b_hh = (const float*)d_in[4];   // [4H]
    float* out = (float*)d_out;                  // [1,B,H]

    char* ws = (char*)d_ws;
    size_t off = 0;
    auto wsalloc = [&](size_t bytes) -> void* {
        void* p = ws + off;
        off = (off + bytes + 255) & ~(size_t)255;
        return p;
    };

    float* hidden   = (float*)wsalloc((size_t)T_DIM * BATCH * H_HID * 4); // 134 MB
    f16*   Xh       = (f16*)  wsalloc((size_t)BATCH * T_DIM * D_DATA * 2); // 67 MB
    f16*   Wihh     = (f16*)  wsalloc((size_t)G4H * D_DATA * 2);
    f16*   Whhh     = (f16*)  wsalloc((size_t)G4H * H_HID * 2);
    float* bias     = (float*)wsalloc((size_t)G4H * 4);
    float* c        = (float*)wsalloc((size_t)BATCH * H_HID * 4);
    f16*   hA       = (f16*)  wsalloc((size_t)BATCH * H_HID * 2);
    f16*   hB       = (f16*)  wsalloc((size_t)BATCH * H_HID * 2);
    float* partials = (float*)wsalloc((size_t)T_DIM * NB_RED * 2 * 4);
    float* dots     = (float*)wsalloc((size_t)T_DIM * 4);
    float* sqs      = (float*)wsalloc((size_t)T_DIM * 4);
    float* alpha    = (float*)wsalloc((size_t)T_DIM * 4);

    // f16 conversions (one-shot, bandwidth-trivial vs the recurrent GEMMs)
    {
        int n = BATCH * T_DIM * D_DATA;
        cvt_f16_kernel<<<(n + 255) / 256, 256, 0, stream>>>(X, Xh, n);
        n = G4H * D_DATA;
        cvt_f16_kernel<<<(n + 255) / 256, 256, 0, stream>>>(W_ih, Wihh, n);
        n = G4H * H_HID;
        cvt_f16_kernel<<<(n + 255) / 256, 256, 0, stream>>>(W_hh, Whhh, n);
    }

    init_kernel<<<(BATCH * H_HID + 255) / 256, 256, 0, stream>>>(
        b_ih, b_hh, bias, c, hA, hB, hidden);

    // sequential LSTM steps; h double-buffered in f16
    for (int t = 1; t < T_DIM; ++t) {
        const f16* hin  = (t & 1) ? hA : hB;
        f16*       hout = (t & 1) ? hB : hA;
        lstm_step_kernel<<<dim3(BATCH / 32, H_HID / 32), 128, 0, stream>>>(
            Xh, Wihh, Whhh, bias, hin, hout, c, hidden, t);
    }

    // attention: deterministic two-pass reductions + softmax + weighted sum
    attn_partial_kernel<<<dim3(NB_RED, T_DIM), 256, 0, stream>>>(hidden, partials);
    attn_reduce_kernel<<<1, 128, 0, stream>>>(partials, dots, sqs);
    attn_softmax_kernel<<<1, 128, 0, stream>>>(dots, sqs, alpha);
    weighted_sum_kernel<<<(BATCH * H_HID) / 256, 256, 0, stream>>>(hidden, alpha, out);
}
